// TriangleMultiplicativeUpdate_24842090840413
// MI455X (gfx1250) — matrix operand order repaired and checked by reading, not compiled, not run on hardware
//
#include <hip/hip_runtime.h>
#include <hip/hip_bf16.h>

// ---------------------------------------------------------------------------
// Triangle multiplicative update (outgoing), B=1, N=512, c_z=c_h=128, f32 I/O.
// All matmuls via v_wmma_f32_16x16x32_bf16; triangle GEMM stages operands
// through LDS with global_load_async_to_lds_b128 (ASYNCcnt) double-buffering.
// ---------------------------------------------------------------------------

#define N       512
#define NN      (512 * 512)          // 262144 rows
#define C       128

typedef __attribute__((ext_vector_type(16))) __bf16 v16bf;
typedef __attribute__((ext_vector_type(8)))  __bf16 v8bf;
typedef __attribute__((ext_vector_type(8)))  float  v8f;

// ---- workspace layout (bytes) ----
#define OFF_W   0ull                              // 6 * 128*128 bf16 weights
#define OFF_ZN  (1ull << 20)                      // zn  : NN*128 bf16 = 64 MB
#define OFF_A   (OFF_ZN + 67108864ull)            // aT  : [c][i][k] bf16 64 MB
#define OFF_B   (OFF_A  + 67108864ull)            // bT  : [c][j][k] bf16 64 MB
#define OFF_G   (OFF_B  + 67108864ull)            // g   : [r][c]    bf16 64 MB
#define OFF_X   (OFF_G  + 67108864ull)            // x   : [c][i][j] f32 128 MB

// ---------------- helpers ----------------
static __device__ __forceinline__ unsigned short f2bfu(float f) {
    unsigned u = __builtin_bit_cast(unsigned, f);
    u += 0x7FFFu + ((u >> 16) & 1u);              // round to nearest even
    return (unsigned short)(u >> 16);
}
static __device__ __forceinline__ __bf16 f2bf(float f) {
    unsigned short s = f2bfu(f);
    return __builtin_bit_cast(__bf16, s);
}
static __device__ __forceinline__ float bf2f(__bf16 h) {
    unsigned u = ((unsigned)__builtin_bit_cast(unsigned short, h)) << 16;
    return __builtin_bit_cast(float, u);
}
static __device__ __forceinline__ float sigm(float x) {
    return 1.0f / (1.0f + __expf(-x));
}
static __device__ __forceinline__ float wsum32(float v) {
#pragma unroll
    for (int m = 16; m >= 1; m >>= 1) v += __shfl_xor(v, m, 32);
    return v;
}

// A fragment: 16x32 bf16, row-major source, lane L<16 holds row M=L K0..15;
// lane L+16 holds K16..31 (contiguous, matching load_b_frag).
static __device__ __forceinline__ v16bf load_a_frag(const __bf16* base, int ld,
                                                    int kb, int lane) {
    const int g = lane >> 4, m = lane & 15;
    return *(const v16bf*)(base + (size_t)m * ld + kb + g * 16);
}

// B fragment: 32x16 bf16 where B_op[k][n]; memory laid out as [n][k]
// (k contiguous). Lane n%16 in group lane/16 reads 16 consecutive k.
static __device__ __forceinline__ v16bf load_b_frag(const __bf16* base, int ld,
                                                    int nb, int kb, int lane) {
    const int kg = lane >> 4, n = lane & 15;
    return *(const v16bf*)(base + (size_t)(nb + n) * ld + kb + kg * 16);
}

static __device__ __forceinline__ v8f wmma_bf16(v16bf a, v16bf b, v8f c) {
    return __builtin_amdgcn_wmma_f32_16x16x32_bf16(false, a, false, b,
                                                   (short)0, c, false, false);
}

// async 16-byte global -> LDS copy (ASYNCcnt) ------------------------------
static __device__ __forceinline__ void async_cp16(const __bf16* g, __bf16* l) {
    unsigned la = (unsigned)(uintptr_t)l;                  // LDS byte offset
    unsigned long long ga = (unsigned long long)(uintptr_t)g;
    asm volatile("global_load_async_to_lds_b128 %0, %1, off"
                 :: "v"(la), "v"(ga) : "memory");
}
static __device__ __forceinline__ void async_wait0() {
    asm volatile("s_wait_asynccnt 0x0" ::: "memory");
}

// ---------------- kernel 0: weights f32 -> bf16 ----------------
__global__ void k_wconv(const float* s0, const float* s1, const float* s2,
                        const float* s3, const float* s4, const float* s5,
                        __bf16* dst) {
    int i = blockIdx.x * blockDim.x + threadIdx.x;      // 6*16384 total
    const float* srcs[6] = {s0, s1, s2, s3, s4, s5};
    int m = i >> 14, off = i & 16383;
    dst[i] = f2bf(srcs[m][off]);
}

// ---------------- kernel 1: LayerNorm(z) -> zn (bf16) ----------------
__global__ void k_ln_in(const float* __restrict__ z,
                        const float* __restrict__ w,
                        const float* __restrict__ b,
                        __bf16* __restrict__ zn) {
    const int lane = threadIdx.x & 31;
    const int wv   = threadIdx.x >> 5;
    const size_t r = (size_t)blockIdx.x * 8 + wv;       // one wave per row
    float4 v = *(const float4*)(z + r * C + lane * 4);
    float mu = wsum32(v.x + v.y + v.z + v.w) * (1.0f / C);
    float dx = v.x - mu, dy = v.y - mu, dz = v.z - mu, dw = v.w - mu;
    float var = wsum32(dx * dx + dy * dy + dz * dz + dw * dw) * (1.0f / C);
    float rs = rsqrtf(var + 1e-5f);
    float4 wc = *(const float4*)(w + lane * 4);
    float4 bc = *(const float4*)(b + lane * 4);
    uint2 pk;
    pk.x = (unsigned)f2bfu(dx * rs * wc.x + bc.x) |
           ((unsigned)f2bfu(dy * rs * wc.y + bc.y) << 16);
    pk.y = (unsigned)f2bfu(dz * rs * wc.z + bc.z) |
           ((unsigned)f2bfu(dw * rs * wc.w + bc.w) << 16);
    *(uint2*)(zn + r * C + lane * 4) = pk;
}

// ---------------- kernel 2: fused 5-way projection + gating ----------------
// waves 0..4 each compute one 16x128 GEMM tile (zn @ W^T) via WMMA.
__global__ void k_proj(const __bf16* __restrict__ zn,
                       const float* __restrict__ mask,
                       const __bf16* __restrict__ wsw,   // [5][128][128] bf16
                       const float* __restrict__ apb, const float* __restrict__ agb,
                       const float* __restrict__ bpb, const float* __restrict__ bgb,
                       const float* __restrict__ ggb,
                       __bf16* __restrict__ wsa,         // [c][NN]
                       __bf16* __restrict__ wsb,         // [c][NN]
                       __bf16* __restrict__ wsg) {       // [NN][c]
    __shared__ float raw[5][16][C];                      // 40 KB
    const int t = threadIdx.x, lane = t & 31, wv = t >> 5;
    const int r0 = blockIdx.x * 16;

    if (wv < 5) {
        const __bf16* W = wsw + (size_t)wv * C * C;
        v8f acc[8];
#pragma unroll
        for (int ob = 0; ob < 8; ++ob) acc[ob] = (v8f){};
#pragma unroll
        for (int kb = 0; kb < C; kb += 32) {
            v16bf va = load_a_frag(zn + (size_t)r0 * C, C, kb, lane);
#pragma unroll
            for (int ob = 0; ob < 8; ++ob) {
                v16bf vb = load_b_frag(W, C, ob * 16, kb, lane);
                acc[ob] = wmma_bf16(va, vb, acc[ob]);
            }
        }
        const int g = lane >> 4, n = lane & 15;
#pragma unroll
        for (int ob = 0; ob < 8; ++ob)
#pragma unroll
            for (int rr = 0; rr < 8; ++rr)
                raw[wv][rr + 8 * g][ob * 16 + n] = acc[ob][rr];
    }
    __syncthreads();

    // a / b epilogue: rows fastest across threads (coalesced channel-major).
#pragma unroll
    for (int it = 0; it < 8; ++it) {
        int idx = it * 256 + t;
        int c = idx >> 4, rr = idx & 15;
        size_t r = (size_t)r0 + rr;
        float mval = mask[r];
        float av = mval * sigm(raw[1][rr][c] + agb[c]) * (raw[0][rr][c] + apb[c]);
        float bv = mval * sigm(raw[3][rr][c] + bgb[c]) * (raw[2][rr][c] + bpb[c]);
        wsa[(size_t)c * NN + r] = f2bf(av);
        wsb[(size_t)c * NN + r] = f2bf(bv);
    }
    // g epilogue: channels fastest (coalesced natural-major).
#pragma unroll
    for (int it = 0; it < 8; ++it) {
        int idx = it * 256 + t;
        int c = idx & 127, rr = idx >> 7;
        wsg[((size_t)r0 + rr) * C + c] = f2bf(sigm(raw[4][rr][c] + ggb[c]));
    }
}

// ---------------- kernel 3: triangle einsum (128 batched GEMMs) ----------------
// X_c = A_c * B_c^T ; A=[c][i][k], B=[c][j][k]. 128x128 block tile; each of 8
// waves owns a 16x128 strip (8 WMMA / K-step). Operands double-buffered in LDS
// via global_load_async_to_lds_b128 (ASYNCcnt).
__global__ void k_tri(const __bf16* __restrict__ wsa,
                      const __bf16* __restrict__ wsb,
                      float* __restrict__ wsx) {
    __shared__ __bf16 sA[2][128][32];                    // 2 x 8 KB
    __shared__ __bf16 sB[2][128][32];                    // 2 x 8 KB  ([j][k])
    const int c = blockIdx.z;
    const int i0 = blockIdx.y * 128, j0 = blockIdx.x * 128;
    const int t = threadIdx.x, lane = t & 31, wv = t >> 5;

    // each thread stages four 16B chunks per K-slab: 2 for A, 2 for B
    const int q0 = t, q1 = t + 256;
    const int ra0 = q0 >> 2, ca0 = (q0 & 3) * 8;
    const int ra1 = q1 >> 2, ca1 = (q1 & 3) * 8;
    const __bf16* Ag0 = wsa + (size_t)c * NN + (size_t)(i0 + ra0) * N + ca0;
    const __bf16* Ag1 = wsa + (size_t)c * NN + (size_t)(i0 + ra1) * N + ca1;
    const __bf16* Bg0 = wsb + (size_t)c * NN + (size_t)(j0 + ra0) * N + ca0;
    const __bf16* Bg1 = wsb + (size_t)c * NN + (size_t)(j0 + ra1) * N + ca1;

    // preload K-slab 0 into buffer 0
    async_cp16(Ag0, &sA[0][ra0][ca0]);
    async_cp16(Ag1, &sA[0][ra1][ca1]);
    async_cp16(Bg0, &sB[0][ra0][ca0]);
    async_cp16(Bg1, &sB[0][ra1][ca1]);
    async_wait0();
    __syncthreads();

    v8f acc[8];
#pragma unroll
    for (int ct = 0; ct < 8; ++ct) acc[ct] = (v8f){};

    int buf = 0;
    for (int kb = 0; kb < N; kb += 32) {
        const int nb = buf ^ 1;
        if (kb + 32 < N) {                               // prefetch next slab
            async_cp16(Ag0 + kb + 32, &sA[nb][ra0][ca0]);
            async_cp16(Ag1 + kb + 32, &sA[nb][ra1][ca1]);
            async_cp16(Bg0 + kb + 32, &sB[nb][ra0][ca0]);
            async_cp16(Bg1 + kb + 32, &sB[nb][ra1][ca1]);
        }
        v16bf va = load_a_frag(&sA[buf][wv * 16][0], 32, 0, lane);
#pragma unroll
        for (int ct = 0; ct < 8; ++ct) {
            v16bf vb = load_b_frag(&sB[buf][0][0], 32, ct * 16, 0, lane);
            acc[ct] = wmma_bf16(va, vb, acc[ct]);
        }
        async_wait0();                                   // next slab landed
        __syncthreads();
        buf = nb;
    }

    const int g = lane >> 4, n = lane & 15;
    float* X = wsx + (size_t)c * NN;
#pragma unroll
    for (int ct = 0; ct < 8; ++ct)
#pragma unroll
        for (int rr = 0; rr < 8; ++rr) {
            size_t rowo = (size_t)(i0 + wv * 16 + rr + 8 * g) * N;
            X[rowo + j0 + ct * 16 + n] = acc[ct][rr];
        }
}

// ---------------- kernel 4: LN(x) + out-projection + gate ----------------
__global__ void k_out(const float* __restrict__ wsx,     // [c][NN]
                      const __bf16* __restrict__ wsg,    // [NN][c]
                      const __bf16* __restrict__ Wz,     // z_w bf16 [128][128]
                      const float* __restrict__ now,
                      const float* __restrict__ nob,
                      const float* __restrict__ zb,
                      float* __restrict__ out) {
    __shared__ float  xs[32][C];                         // 16 KB
    __shared__ __bf16 zs[32][C];                         // 8 KB
    const int t = threadIdx.x, lane = t & 31, wv = t >> 5;
    const size_t r0 = (size_t)blockIdx.x * 32;

    // transpose-gather x: [c][r] -> LDS [r][c], coalesced along r
#pragma unroll
    for (int it = 0; it < 16; ++it) {
        int idx = it * 256 + t;
        int c = idx >> 5, rr = idx & 31;
        xs[rr][c] = wsx[(size_t)c * NN + r0 + rr];
    }
    __syncthreads();

    // LayerNorm over c: wave wv handles rows wv*4 .. wv*4+3
#pragma unroll
    for (int u = 0; u < 4; ++u) {
        int rr = wv * 4 + u;
        float4 v = *(const float4*)(&xs[rr][lane * 4]);
        float mu = wsum32(v.x + v.y + v.z + v.w) * (1.0f / C);
        float dx = v.x - mu, dy = v.y - mu, dz = v.z - mu, dw = v.w - mu;
        float var = wsum32(dx * dx + dy * dy + dz * dz + dw * dw) * (1.0f / C);
        float rs = rsqrtf(var + 1e-5f);
        float4 wc = *(const float4*)(now + lane * 4);
        float4 bc = *(const float4*)(nob + lane * 4);
        uint2 pk;
        pk.x = (unsigned)f2bfu(dx * rs * wc.x + bc.x) |
               ((unsigned)f2bfu(dy * rs * wc.y + bc.y) << 16);
        pk.y = (unsigned)f2bfu(dz * rs * wc.z + bc.z) |
               ((unsigned)f2bfu(dw * rs * wc.w + bc.w) << 16);
        *(uint2*)(&zs[rr][lane * 4]) = pk;
    }
    __syncthreads();

    // out-projection GEMM via WMMA: wave = (row-tile rt, 32-col slab cw)
    const int rt = wv & 1, cw = wv >> 1;
    v8f acc[2] = {(v8f){}, (v8f){}};
#pragma unroll
    for (int kb = 0; kb < C; kb += 32) {
        v16bf va = load_a_frag(&zs[rt * 16][0], C, kb, lane);
#pragma unroll
        for (int ct = 0; ct < 2; ++ct) {
            v16bf vb = load_b_frag(Wz, C, cw * 32 + ct * 16, kb, lane);
            acc[ct] = wmma_bf16(va, vb, acc[ct]);
        }
    }
    const int g = lane >> 4, n = lane & 15;
#pragma unroll
    for (int ct = 0; ct < 2; ++ct) {
        int col = cw * 32 + ct * 16 + n;
        float bias = zb[col];
#pragma unroll
        for (int rr = 0; rr < 8; ++rr) {
            size_t r = r0 + rt * 16 + rr + 8 * g;
            float gate = bf2f(wsg[r * C + col]);
            out[r * C + col] = (acc[ct][rr] + bias) * gate;
        }
    }
}

// ---------------- host launcher ----------------
extern "C" void kernel_launch(void* const* d_in, const int* in_sizes, int n_in,
                              void* d_out, int out_size, void* d_ws, size_t ws_size,
                              hipStream_t stream) {
    const float* z    = (const float*)d_in[0];
    const float* mask = (const float*)d_in[1];
    const float* niw  = (const float*)d_in[2];
    const float* nib  = (const float*)d_in[3];
    const float* apw  = (const float*)d_in[4];
    const float* apb  = (const float*)d_in[5];
    const float* agw  = (const float*)d_in[6];
    const float* agb  = (const float*)d_in[7];
    const float* bpw  = (const float*)d_in[8];
    const float* bpb  = (const float*)d_in[9];
    const float* bgw  = (const float*)d_in[10];
    const float* bgb  = (const float*)d_in[11];
    const float* gw   = (const float*)d_in[12];
    const float* gbi  = (const float*)d_in[13];
    const float* zw   = (const float*)d_in[14];
    const float* zb   = (const float*)d_in[15];
    const float* now  = (const float*)d_in[16];
    const float* nob  = (const float*)d_in[17];
    float* out = (float*)d_out;

    char* ws = (char*)d_ws;
    __bf16* wsw = (__bf16*)(ws + OFF_W);
    __bf16* zn  = (__bf16*)(ws + OFF_ZN);
    __bf16* wsa = (__bf16*)(ws + OFF_A);
    __bf16* wsb = (__bf16*)(ws + OFF_B);
    __bf16* wsg = (__bf16*)(ws + OFF_G);
    float*  wsx = (float*)(ws + OFF_X);

    k_wconv<<<(6 * C * C) / 256, 256, 0, stream>>>(apw, agw, bpw, bgw, gw, zw, wsw);
    k_ln_in<<<NN / 8, 256, 0, stream>>>(z, niw, nib, zn);
    k_proj<<<NN / 16, 256, 0, stream>>>(zn, mask, wsw, apb, agb, bpb, bgb, gbi,
                                        wsa, wsb, wsg);
    k_tri<<<dim3(4, 4, 128), 256, 0, stream>>>(wsa, wsb, wsx);
    k_out<<<NN / 32, 256, 0, stream>>>(wsx, wsg, wsw + 5 * C * C, now, nob, zb, out);
}
